// GREATEncoder_83434034692204
// MI455X (gfx1250) — matrix operand order repaired and checked by reading, not compiled, not run on hardware
//
#include <hip/hip_runtime.h>
#include <cstdint>
#include <cstddef>

// ---------------- problem constants (fixed by the reference) ----------------
#define NN      384                 // nodes
#define HH      128                 // embedding dim
#define NHEADS  8
#define LL      3
#define FFH     512                 // ff hidden
#define NE      (NN*(NN-1))         // 147072 edges
#define QKVN    (6*HH)              // 768
#define EPSV    1e-5f
#define SCALEV  0.25f               // 1/sqrt(16)

typedef __attribute__((ext_vector_type(16))) __bf16 v16bf;
typedef __attribute__((ext_vector_type(8)))  __bf16 v8bf;
typedef __attribute__((ext_vector_type(4)))  __bf16 v4bf;
typedef __attribute__((ext_vector_type(8)))  float  v8f;

// triu pair index helpers: pair (i,j), i<j has t = pair_off(i)+(j-i-1);
// edge 2t = (i->j), edge 2t+1 = (j->i).
__device__ __forceinline__ int pair_off(int i) { return i * (2 * NN - 1 - i) / 2; }
// directed edge id for (i -> n), i != n
__device__ __forceinline__ int edge_in(int i, int n) {
  return (i < n) ? 2 * (pair_off(i) + (n - i - 1))
                 : 2 * (pair_off(n) + (i - n - 1)) + 1;
}

// CDNA5 async global->LDS copy (16B per lane), tracked by ASYNCcnt.
__device__ __forceinline__ void great_async_b128(unsigned lds, const void* g) {
  asm volatile("global_load_async_to_lds_b128 %0, %1, off"
               :: "v"(lds), "v"(g) : "memory");
}
__device__ __forceinline__ void great_wait_async0() {
  asm volatile("s_wait_asynccnt 0x0" ::: "memory");
}

// ---------------------------------------------------------------------------
// embed: x[e,c] = edge_attr[e,:3] @ embed_W[:3,c] + embed_b[c]
// ---------------------------------------------------------------------------
__global__ __launch_bounds__(256) void great_embed(
    const float* __restrict__ ea, const float* __restrict__ W,
    const float* __restrict__ b, float* __restrict__ x) {
  int gid = blockIdx.x * 256 + threadIdx.x;       // NE*128 threads
  int e = gid >> 7, c = gid & 127;
  float a0 = ea[e * 3 + 0], a1 = ea[e * 3 + 1], a2 = ea[e * 3 + 2];
  x[gid] = fmaf(a0, W[c], fmaf(a1, W[128 + c], fmaf(a2, W[256 + c], b[c])));
}

// ---------------------------------------------------------------------------
// repack qkv_W[l] : [6][128][128] -> Wq [128][768] row-major (n = k*128+h)
// ---------------------------------------------------------------------------
__global__ __launch_bounds__(256) void great_repack_qkv(
    const float* __restrict__ src, float* __restrict__ Wq) {
  int gid = blockIdx.x * 256 + threadIdx.x;       // 6*128*128 = 98304
  int k = gid >> 14, d = (gid >> 7) & 127, h = gid & 127;
  Wq[d * QKVN + k * 128 + h] = src[gid];
}

// ---------------------------------------------------------------------------
// generic bf16-WMMA GEMM:  C[M,Ncol] = act(A[M,K] @ B[K,Ncol] + bias)
//   A: fp32 (converted while staging) or bf16 (async-copied straight to LDS)
//   B: fp32 in global, converted to bf16 and stored transposed in LDS
// block: 256 thr = 8 waves; block tile 64(M) x 64(N); wave tile 16 x 32;
// K-step 32; double-buffered LDS, one barrier per K-step.
// fragment K order: contiguous K = 16*hi + i in both A and B (wave32).
// ---------------------------------------------------------------------------
#define LDSB 40   // padded bf16 row stride (80 bytes: 16B aligned, conflict-free)

template <bool A_IS_BF16, bool RELU, bool OUT_BF16>
__global__ __launch_bounds__(256) void great_gemm(
    const void* __restrict__ Ap, const float* __restrict__ B,
    const float* __restrict__ bias, void* __restrict__ Cp,
    int M, int K, int Ncol) {
  __shared__ __attribute__((aligned(16))) __bf16 As[2][64 * LDSB];
  __shared__ __attribute__((aligned(16))) __bf16 Bs[2][64 * LDSB];

  const int tid  = threadIdx.x;
  const int m0   = blockIdx.y * 64;
  const int n0   = blockIdx.x * 64;
  const int w    = tid >> 5, lane = tid & 31;
  const int wm   = w >> 1,  wn   = w & 1;          // 4 x 2 wave grid
  const int r    = lane & 15, hi = lane >> 4;

  // A staging: thread owns 8 contiguous K of one row (16B bf16)
  const int arow = tid >> 2;                       // 0..63
  const int acg  = (tid & 3) * 8;                  // 0,8,16,24
  // B staging: thread owns a 4(K) x 2(N) micro-tile
  const int bkq  = (tid >> 5) * 4;                 // k base: 0,4,..,28
  const int bnp  = (tid & 31) * 2;                 // n base: 0..62 step 2

  v8f acc0 = {}; v8f acc1 = {};

  const int T = K >> 5;

  // ---- prologue: stage tile 0 into buffer 0 ----
  if (A_IS_BF16) {
    const __bf16* g = (const __bf16*)Ap + (size_t)(m0 + arow) * K + acg;
    great_async_b128((unsigned)(uintptr_t)&As[0][arow * LDSB + acg], g);
  } else {
    const float* g = (const float*)Ap + (size_t)(m0 + arow) * K + acg;
    v8bf av;
#pragma unroll
    for (int i = 0; i < 8; ++i) av[i] = (__bf16)g[i];
    *reinterpret_cast<v8bf*>(&As[0][arow * LDSB + acg]) = av;
  }
  {
    const float* g = B + (size_t)bkq * Ncol + n0 + bnp;
    v4bf c0, c1;
#pragma unroll
    for (int j = 0; j < 4; ++j) {
      float2 f = *reinterpret_cast<const float2*>(g + (size_t)j * Ncol);
      c0[j] = (__bf16)f.x; c1[j] = (__bf16)f.y;
    }
    *reinterpret_cast<v4bf*>(&Bs[0][bnp * LDSB + bkq]) = c0;
    *reinterpret_cast<v4bf*>(&Bs[0][(bnp + 1) * LDSB + bkq]) = c1;
  }

  for (int t = 0; t < T; ++t) {
    const int cur = t & 1, nxt = cur ^ 1;
    if (A_IS_BF16) great_wait_async0();   // our async writes to buf[cur] landed
    __syncthreads();

    // ---- issue global loads for tile t+1 (overlap with WMMA below) ----
    const bool have_next = (t + 1) < T;
    float  a_next[8];
    float2 b_next[4];
    if (have_next) {
      const int k0 = (t + 1) * 32;
      if (A_IS_BF16) {
        const __bf16* g = (const __bf16*)Ap + (size_t)(m0 + arow) * K + k0 + acg;
        great_async_b128((unsigned)(uintptr_t)&As[nxt][arow * LDSB + acg], g);
      } else {
        const float* g = (const float*)Ap + (size_t)(m0 + arow) * K + k0 + acg;
#pragma unroll
        for (int i = 0; i < 8; ++i) a_next[i] = g[i];
      }
      const float* gb = B + (size_t)(k0 + bkq) * Ncol + n0 + bnp;
#pragma unroll
      for (int j = 0; j < 4; ++j)
        b_next[j] = *reinterpret_cast<const float2*>(gb + (size_t)j * Ncol);
      if (t + 2 < T) {  // L2 prefetch for tile t+2 (global_prefetch_b8)
        const int k2 = (t + 2) * 32;
        if (!A_IS_BF16)
          __builtin_prefetch((const float*)Ap + (size_t)(m0 + arow) * K + k2 + acg, 0, 1);
        __builtin_prefetch(B + (size_t)(k2 + bkq) * Ncol + n0 + bnp, 0, 1);
      }
    }

    // ---- A fragment: lane holds row (wm*16+r), K = 16*hi .. 16*hi+15 ----
    v16bf af;
    {
      v8bf alo = *reinterpret_cast<const v8bf*>(&As[cur][(wm * 16 + r) * LDSB + 16 * hi]);
      v8bf ahi = *reinterpret_cast<const v8bf*>(&As[cur][(wm * 16 + r) * LDSB + 16 * hi + 8]);
#pragma unroll
      for (int i = 0; i < 8; ++i) { af[i] = alo[i]; af[8 + i] = ahi[i]; }
    }
    // ---- B fragments: lane holds col (nb+r), K = 16*hi .. 16*hi+15 ----
    v16bf bf0, bf1;
    {
      v8bf b0a = *reinterpret_cast<const v8bf*>(&Bs[cur][(wn * 32 + r) * LDSB + 16 * hi]);
      v8bf b0b = *reinterpret_cast<const v8bf*>(&Bs[cur][(wn * 32 + r) * LDSB + 16 * hi + 8]);
      v8bf b1a = *reinterpret_cast<const v8bf*>(&Bs[cur][(wn * 32 + 16 + r) * LDSB + 16 * hi]);
      v8bf b1b = *reinterpret_cast<const v8bf*>(&Bs[cur][(wn * 32 + 16 + r) * LDSB + 16 * hi + 8]);
#pragma unroll
      for (int i = 0; i < 8; ++i) {
        bf0[i] = b0a[i]; bf0[8 + i] = b0b[i];
        bf1[i] = b1a[i]; bf1[8 + i] = b1b[i];
      }
    }
    acc0 = __builtin_amdgcn_wmma_f32_16x16x32_bf16(false, af, false, bf0,
                                                   (short)0, acc0, false, false);
    acc1 = __builtin_amdgcn_wmma_f32_16x16x32_bf16(false, af, false, bf1,
                                                   (short)0, acc1, false, false);

    // ---- convert + store tile t+1 into the other buffer ----
    if (have_next) {
      if (!A_IS_BF16) {
        v8bf av;
#pragma unroll
        for (int i = 0; i < 8; ++i) av[i] = (__bf16)a_next[i];
        *reinterpret_cast<v8bf*>(&As[nxt][arow * LDSB + acg]) = av;
      }
      v4bf c0, c1;
#pragma unroll
      for (int j = 0; j < 4; ++j) { c0[j] = (__bf16)b_next[j].x; c1[j] = (__bf16)b_next[j].y; }
      *reinterpret_cast<v4bf*>(&Bs[nxt][bnp * LDSB + bkq]) = c0;
      *reinterpret_cast<v4bf*>(&Bs[nxt][(bnp + 1) * LDSB + bkq]) = c1;
    }
  }

  // ---- epilogue: D lane layout: n = lane&15, VGPR i -> M = i + 8*hi ----
  const int col0 = n0 + wn * 32 + r;
  const int col1 = col0 + 16;
  const float bv0 = bias[col0];
  const float bv1 = bias[col1];
#pragma unroll
  for (int i = 0; i < 8; ++i) {
    const int row = m0 + wm * 16 + i + 8 * hi;
    float v0 = acc0[i] + bv0;
    float v1 = acc1[i] + bv1;
    if (RELU) { v0 = fmaxf(v0, 0.f); v1 = fmaxf(v1, 0.f); }
    if (OUT_BF16) {
      __bf16* C = (__bf16*)Cp;
      C[(size_t)row * Ncol + col0] = (__bf16)v0;
      C[(size_t)row * Ncol + col1] = (__bf16)v1;
    } else {
      float* C = (float*)Cp;
      C[(size_t)row * Ncol + col0] = v0;
      C[(size_t)row * Ncol + col1] = v1;
    }
  }
}

// ---------------------------------------------------------------------------
// attention scores: s_in[e,h] = (q_in . k_in)*SCALE, s_out likewise.
// q/k live in the projection buffer p[E,768] bf16:
//   q_in +0, k_in +128, v_in +256, q_out +384, k_out +512, v_out +640
// ---------------------------------------------------------------------------
__global__ __launch_bounds__(256) void great_score(
    const __bf16* __restrict__ p, float* __restrict__ s_in,
    float* __restrict__ s_out) {
  int gid = blockIdx.x * 256 + threadIdx.x;       // NE*8 threads
  int e = gid >> 3, h = gid & 7;
  const __bf16* row = p + (size_t)e * QKVN + h * 16;
  const v8bf* qi = reinterpret_cast<const v8bf*>(row);
  const v8bf* ki = reinterpret_cast<const v8bf*>(row + 128);
  const v8bf* qo = reinterpret_cast<const v8bf*>(row + 384);
  const v8bf* ko = reinterpret_cast<const v8bf*>(row + 512);
  v8bf qi0 = qi[0], qi1 = qi[1], ki0 = ki[0], ki1 = ki[1];
  v8bf qo0 = qo[0], qo1 = qo[1], ko0 = ko[0], ko1 = ko[1];
  float si = 0.f, so = 0.f;
#pragma unroll
  for (int i = 0; i < 8; ++i) {
    si += (float)qi0[i] * (float)ki0[i] + (float)qi1[i] * (float)ki1[i];
    so += (float)qo0[i] * (float)ko0[i] + (float)qo1[i] * (float)ko1[i];
  }
  s_in[gid] = si * SCALEV;
  s_out[gid] = so * SCALEV;
}

// ---------------------------------------------------------------------------
// segment softmax reduction, node-centric (no atomics):
// per node n: max & sum(exp) over incoming edges (in, grouped by dst) and
// outgoing edges (out, grouped by src). e_out = e_in ^ 1 (pair-mates).
// ---------------------------------------------------------------------------
__global__ __launch_bounds__(128) void great_softmax_reduce(
    const float* __restrict__ s_in, const float* __restrict__ s_out,
    float* __restrict__ Min, float* __restrict__ Sin,
    float* __restrict__ Mout, float* __restrict__ Sout) {
  __shared__ float red[128 * 8];
  __shared__ float shMin[8], shMout[8];
  const int n = blockIdx.x, tid = threadIdx.x;

  float mi[8], mo[8];
#pragma unroll
  for (int h = 0; h < 8; ++h) { mi[h] = -1e30f; mo[h] = -1e30f; }
  for (int ii = tid; ii < NN - 1; ii += 128) {
    int i = (ii < n) ? ii : ii + 1;
    int ein = edge_in(i, n);
    int eout = ein ^ 1;
#pragma unroll
    for (int h = 0; h < 8; ++h) {
      mi[h] = fmaxf(mi[h], s_in[(size_t)ein * 8 + h]);
      mo[h] = fmaxf(mo[h], s_out[(size_t)eout * 8 + h]);
    }
  }
#pragma unroll
  for (int h = 0; h < 8; ++h) red[tid * 8 + h] = mi[h];
  __syncthreads();
  if (tid < 8) {
    float m = -1e30f;
    for (int j = 0; j < 128; ++j) m = fmaxf(m, red[j * 8 + tid]);
    shMin[tid] = m;
  }
  __syncthreads();
#pragma unroll
  for (int h = 0; h < 8; ++h) red[tid * 8 + h] = mo[h];
  __syncthreads();
  if (tid < 8) {
    float m = -1e30f;
    for (int j = 0; j < 128; ++j) m = fmaxf(m, red[j * 8 + tid]);
    shMout[tid] = m;
  }
  __syncthreads();

  float si[8], so[8];
#pragma unroll
  for (int h = 0; h < 8; ++h) { si[h] = 0.f; so[h] = 0.f; }
  for (int ii = tid; ii < NN - 1; ii += 128) {
    int i = (ii < n) ? ii : ii + 1;
    int ein = edge_in(i, n);
    int eout = ein ^ 1;
#pragma unroll
    for (int h = 0; h < 8; ++h) {
      si[h] += expf(s_in[(size_t)ein * 8 + h] - shMin[h]);
      so[h] += expf(s_out[(size_t)eout * 8 + h] - shMout[h]);
    }
  }
#pragma unroll
  for (int h = 0; h < 8; ++h) red[tid * 8 + h] = si[h];
  __syncthreads();
  if (tid < 8) {
    float s = 0.f;
    for (int j = 0; j < 128; ++j) s += red[j * 8 + tid];
    Min[n * 8 + tid] = shMin[tid];
    Sin[n * 8 + tid] = s;
  }
  __syncthreads();
#pragma unroll
  for (int h = 0; h < 8; ++h) red[tid * 8 + h] = so[h];
  __syncthreads();
  if (tid < 8) {
    float s = 0.f;
    for (int j = 0; j < 128; ++j) s += red[j * 8 + tid];
    Mout[n * 8 + tid] = shMout[tid];
    Sout[n * 8 + tid] = s;
  }
}

// ---------------------------------------------------------------------------
// finalize attention weights in place: a = exp(s - M[seg]) / S[seg]
// ---------------------------------------------------------------------------
__global__ __launch_bounds__(256) void great_attn_finalize(
    float* __restrict__ s_in, float* __restrict__ s_out,
    const int* __restrict__ ei, const float* __restrict__ Min,
    const float* __restrict__ Sin, const float* __restrict__ Mout,
    const float* __restrict__ Sout) {
  int gid = blockIdx.x * 256 + threadIdx.x;       // NE*8
  int e = gid >> 3, h = gid & 7;
  int src = ei[e], dst = ei[NE + e];
  s_in[gid]  = expf(s_in[gid]  - Min[dst * 8 + h])  / Sin[dst * 8 + h];
  s_out[gid] = expf(s_out[gid] - Mout[src * 8 + h]) / Sout[src * 8 + h];
}

// ---------------------------------------------------------------------------
// message aggregation, node-centric, no atomics:
// node_msg[n, h, 0:16]=sum a_in[e]*v_in[e], 16:32 -> pair-mate, 32:48 o_out,
// 48:64 pair-mate o_out; over incoming edges e of node n.
// ---------------------------------------------------------------------------
__global__ __launch_bounds__(128) void great_aggregate(
    const float* __restrict__ a_in, const float* __restrict__ a_out,
    const __bf16* __restrict__ p, float* __restrict__ node_msg) {
  const int n = blockIdx.x, tid = threadIdx.x;
  const int h = tid >> 4, d = tid & 15;
  float acc0 = 0.f, acc1 = 0.f, acc2 = 0.f, acc3 = 0.f;
  for (int ii = 0; ii < NN - 1; ++ii) {
    int i = (ii < n) ? ii : ii + 1;
    int e = edge_in(i, n);
    int e2 = e ^ 1;
    float ai  = a_in[(size_t)e * 8 + h];
    float ai2 = a_in[(size_t)e2 * 8 + h];
    float ao  = a_out[(size_t)e * 8 + h];
    float ao2 = a_out[(size_t)e2 * 8 + h];
    acc0 += ai  * (float)p[(size_t)e  * QKVN + 256 + h * 16 + d];
    acc1 += ai2 * (float)p[(size_t)e2 * QKVN + 256 + h * 16 + d];
    acc2 += ao  * (float)p[(size_t)e  * QKVN + 640 + h * 16 + d];
    acc3 += ao2 * (float)p[(size_t)e2 * QKVN + 640 + h * 16 + d];
  }
  float* dstp = node_msg + (size_t)n * 512 + h * 64 + d;
  dstp[0] = acc0; dstp[16] = acc1; dstp[32] = acc2; dstp[48] = acc3;
}

// ---------------------------------------------------------------------------
// agg gather: agg[e, 0:128]=node_out[src[e]], agg[e,128:256]=node_out[dst[e]]
// ---------------------------------------------------------------------------
__global__ __launch_bounds__(256) void great_agg_gather(
    const float* __restrict__ node_out, const int* __restrict__ ei,
    __bf16* __restrict__ agg) {
  int gid = blockIdx.x * 256 + threadIdx.x;       // NE*256
  int e = gid >> 8, c = gid & 255;
  int node = (c < 128) ? ei[e] : ei[NE + e];
  agg[gid] = (__bf16)node_out[node * 128 + (c & 127)];
}

// ---------------------------------------------------------------------------
// batch-norm: stats (sum, sumsq per channel of y = tmp + x) then apply.
// ---------------------------------------------------------------------------
__global__ __launch_bounds__(256) void great_zero(float* __restrict__ ptr, int n) {
  int gid = blockIdx.x * 256 + threadIdx.x;
  if (gid < n) ptr[gid] = 0.f;
}

__global__ __launch_bounds__(128) void great_bn_stats(
    const float* __restrict__ tmp, const float* __restrict__ x,
    float* __restrict__ stats) {
  const int c = threadIdx.x;
  float s = 0.f, q = 0.f;
  for (int r = blockIdx.x; r < NE; r += gridDim.x) {
    float y = tmp[(size_t)r * 128 + c] + x[(size_t)r * 128 + c];
    s += y; q += y * y;
  }
  atomicAdd(&stats[c], s);
  atomicAdd(&stats[128 + c], q);
}

__global__ __launch_bounds__(256) void great_bn_apply(
    const float* __restrict__ tmp, float* __restrict__ x,
    const float* __restrict__ stats, const float* __restrict__ g,
    const float* __restrict__ b) {
  int gid = blockIdx.x * 256 + threadIdx.x;       // NE*128
  int c = gid & 127;
  const float inv = 1.0f / (float)NE;
  float mean = stats[c] * inv;
  float var = stats[128 + c] * inv - mean * mean;
  float y = tmp[gid] + x[gid];
  x[gid] = g[c] * (y - mean) * rsqrtf(var + EPSV) + b[c];
}

// ---------------------------------------------------------------------------
static inline char* carve(char*& base, size_t bytes) {
  char* pp = base;
  base += (bytes + 255) & ~size_t(255);
  return pp;
}

extern "C" void kernel_launch(void* const* d_in, const int* in_sizes, int n_in,
                              void* d_out, int out_size, void* d_ws, size_t ws_size,
                              hipStream_t stream) {
  (void)in_sizes; (void)n_in; (void)out_size; (void)ws_size;

  const float* edge_attr = (const float*)d_in[0];
  const int*   ei        = (const int*)d_in[1];
  const float* embed_W   = (const float*)d_in[3];
  const float* embed_b   = (const float*)d_in[4];
  const float* qkv_W     = (const float*)d_in[5];
  const float* qkv_b     = (const float*)d_in[6];
  const float* lino_W    = (const float*)d_in[7];
  const float* lino_b    = (const float*)d_in[8];
  const float* edgeo_W   = (const float*)d_in[9];
  const float* edgeo_b   = (const float*)d_in[10];
  const float* ff1_W     = (const float*)d_in[11];
  const float* ff1_b     = (const float*)d_in[12];
  const float* ff2_W     = (const float*)d_in[13];
  const float* ff2_b     = (const float*)d_in[14];
  const float* bn_att_g  = (const float*)d_in[15];
  const float* bn_att_b  = (const float*)d_in[16];
  const float* bn_ff_g   = (const float*)d_in[17];
  const float* bn_ff_b   = (const float*)d_in[18];

  float* x = (float*)d_out;                       // edge state [E,128], final output

  char* base = (char*)d_ws;
  __bf16* p        = (__bf16*)carve(base, (size_t)NE * QKVN * sizeof(__bf16));
  float*  s_in     = (float*)carve(base, (size_t)NE * 8 * sizeof(float));
  float*  s_out    = (float*)carve(base, (size_t)NE * 8 * sizeof(float));
  float*  Min      = (float*)carve(base, (size_t)NN * 8 * sizeof(float));
  float*  Sin      = (float*)carve(base, (size_t)NN * 8 * sizeof(float));
  float*  Mout     = (float*)carve(base, (size_t)NN * 8 * sizeof(float));
  float*  Sout     = (float*)carve(base, (size_t)NN * 8 * sizeof(float));
  float*  node_msg = (float*)carve(base, (size_t)NN * 512 * sizeof(float));
  float*  node_out = (float*)carve(base, (size_t)NN * 128 * sizeof(float));
  __bf16* agg      = (__bf16*)carve(base, (size_t)NE * 256 * sizeof(__bf16));
  float*  tmp      = (float*)carve(base, (size_t)NE * 128 * sizeof(float));
  __bf16* hbuf     = (__bf16*)carve(base, (size_t)NE * FFH * sizeof(__bf16));
  float*  Wq       = (float*)carve(base, (size_t)HH * QKVN * sizeof(float));
  float*  stats    = (float*)carve(base, 256 * sizeof(float));

  // ---- embedding ----
  great_embed<<<NE * 128 / 256, 256, 0, stream>>>(edge_attr, embed_W, embed_b, x);

  for (int l = 0; l < LL; ++l) {
    // QKV projection: p[E,768] = x @ Wq + qkv_b[l]   (bf16 out)
    great_repack_qkv<<<6 * 128 * 128 / 256, 256, 0, stream>>>(qkv_W + (size_t)l * 6 * HH * HH, Wq);
    great_gemm<false, false, true><<<dim3(QKVN / 64, NE / 64), 256, 0, stream>>>(
        x, Wq, qkv_b + (size_t)l * QKVN, p, NE, HH, QKVN);

    // attention
    great_score<<<NE * 8 / 256, 256, 0, stream>>>(p, s_in, s_out);
    great_softmax_reduce<<<NN, 128, 0, stream>>>(s_in, s_out, Min, Sin, Mout, Sout);
    great_attn_finalize<<<NE * 8 / 256, 256, 0, stream>>>(s_in, s_out, ei, Min, Sin, Mout, Sout);
    great_aggregate<<<NN, 128, 0, stream>>>(s_in, s_out, p, node_msg);

    // node linear: node_out[384,128] = node_msg[384,512] @ lino_W[l] + lino_b[l]
    great_gemm<false, false, false><<<dim3(HH / 64, NN / 64), 256, 0, stream>>>(
        node_msg, lino_W + (size_t)l * 4 * HH * HH, lino_b + (size_t)l * HH,
        node_out, NN, 4 * HH, HH);

    // edge output: tmp[E,128] = concat(node[src],node[dst]) @ edgeo_W[l] + b
    great_agg_gather<<<NE, 256, 0, stream>>>(node_out, ei, agg);
    great_gemm<true, false, false><<<dim3(HH / 64, NE / 64), 256, 0, stream>>>(
        agg, edgeo_W + (size_t)l * 2 * HH * HH, edgeo_b + (size_t)l * HH,
        tmp, NE, 2 * HH, HH);

    // BN(att): x = bn(tmp + x)
    great_zero<<<1, 256, 0, stream>>>(stats, 256);
    great_bn_stats<<<512, 128, 0, stream>>>(tmp, x, stats);
    great_bn_apply<<<NE * 128 / 256, 256, 0, stream>>>(
        tmp, x, stats, bn_att_g + (size_t)l * HH, bn_att_b + (size_t)l * HH);

    // feed-forward: h = relu(x @ ff1_W + b) (bf16), tmp = h @ ff2_W + b (f32)
    great_gemm<false, true, true><<<dim3(FFH / 64, NE / 64), 256, 0, stream>>>(
        x, ff1_W + (size_t)l * HH * FFH, ff1_b + (size_t)l * FFH, hbuf, NE, HH, FFH);
    great_gemm<true, false, false><<<dim3(HH / 64, NE / 64), 256, 0, stream>>>(
        hbuf, ff2_W + (size_t)l * FFH * HH, ff2_b + (size_t)l * HH, tmp, NE, FFH, HH);

    // BN(ff): x = bn(tmp + x)
    great_zero<<<1, 256, 0, stream>>>(stats, 256);
    great_bn_stats<<<512, 128, 0, stream>>>(tmp, x, stats);
    great_bn_apply<<<NE * 128 / 256, 256, 0, stream>>>(
        tmp, x, stats, bn_ff_g + (size_t)l * HH, bn_ff_b + (size_t)l * HH);
  }
}
